// Inter_SA_3367254360475
// MI455X (gfx1250) — compile-verified, matrix operand order checked by reading, not run on hardware
//
#include <hip/hip_runtime.h>
#include <hip/hip_fp16.h>
#include <math.h>

// ---------------------------------------------------------------------------
// Types for CDNA5 WMMA (wave32, 16x16x32 f16 -> f32)
// ---------------------------------------------------------------------------
typedef _Float16 h16;
typedef __attribute__((ext_vector_type(16))) _Float16 v16h;
typedef __attribute__((ext_vector_type(8)))  _Float16 v8h;
typedef __attribute__((ext_vector_type(8)))  float    v8f;

#define TOKS 65536   // B*H*W = 16*64*64
#define CC   256
#define HW   4096    // 64*64

// Fragment layout helper (ISA 7.12.2): element e -> K = e + (e&8) + 8*hi,
// so a 16x32 f16 A/B fragment is two contiguous b128 LDS loads per lane.
__device__ __forceinline__ v16h frag_ld(const h16* rowbase, int hi) {
  v8h lo = *(const v8h*)(rowbase + hi * 8);
  v8h hh = *(const v8h*)(rowbase + 16 + hi * 8);
  return __builtin_shufflevector(lo, hh, 0,1,2,3,4,5,6,7,8,9,10,11,12,13,14,15);
}

__device__ __forceinline__ v8f wmma16(v16h a, v16h b, v8f c) {
  return __builtin_amdgcn_wmma_f32_16x16x32_f16(false, a, false, b, (short)0, c,
                                                false, false);
}

// Async global->LDS copy (16B per lane), tracked by ASYNCcnt.
// lds_byte: LDS byte address (low 32 bits of a generic shared pointer),
// base: uniform 64-bit global base (SGPR pair), goff: per-lane byte offset.
__device__ __forceinline__ void async_b128(uint32_t lds_byte, uint64_t base,
                                           int goff) {
  asm volatile("global_load_async_to_lds_b128 %0, %1, %2"
               :: "v"(lds_byte), "v"(goff), "s"(base) : "memory");
}
__device__ __forceinline__ void wait_async0() {
  asm volatile("s_wait_asynccnt 0" ::: "memory");
}

// ---------------------------------------------------------------------------
// f32 -> f16 weight pre-conversion (once per launch; weights ~1.5MB total)
// ---------------------------------------------------------------------------
__global__ __launch_bounds__(256)
void cvt_f32_to_f16(const float* __restrict__ in, h16* __restrict__ out, int n) {
  int i = blockIdx.x * blockDim.x + threadIdx.x;
  if (i < n) out[i] = (h16)in[i];
}

// ---------------------------------------------------------------------------
// LayerNorm kernels (one thread per token, C=256)
// ---------------------------------------------------------------------------
__global__ __launch_bounds__(256)
void ln_nchw_to_tok16(const float* __restrict__ x, const float* __restrict__ g,
                      const float* __restrict__ b, h16* __restrict__ out) {
  int tok = blockIdx.x * blockDim.x + threadIdx.x;
  int bb = tok >> 12, s = tok & (HW - 1);
  const float* xb = x + (size_t)bb * CC * HW + s;
  float sum = 0.f, sq = 0.f;
  for (int c = 0; c < CC; ++c) { float v = xb[(size_t)c * HW]; sum += v; sq += v * v; }
  float mu = sum * (1.f / CC);
  float var = sq * (1.f / CC) - mu * mu;
  float rs = rsqrtf(var + 1e-5f);
  h16* o = out + (size_t)tok * CC;
  for (int c = 0; c < CC; ++c)
    o[c] = (h16)((xb[(size_t)c * HW] - mu) * rs * g[c] + b[c]);
}

__global__ __launch_bounds__(256)
void ln_tok16(const float* __restrict__ x, const float* __restrict__ g,
              const float* __restrict__ b, h16* __restrict__ out) {
  int tok = blockIdx.x * blockDim.x + threadIdx.x;
  const float* xr = x + (size_t)tok * CC;
  float sum = 0.f, sq = 0.f;
  for (int c = 0; c < CC; ++c) { float v = xr[c]; sum += v; sq += v * v; }
  float mu = sum * (1.f / CC);
  float var = sq * (1.f / CC) - mu * mu;
  float rs = rsqrtf(var + 1e-5f);
  h16* o = out + (size_t)tok * CC;
  for (int c = 0; c < CC; ++c)
    o[c] = (h16)((xr[c] - mu) * rs * g[c] + b[c]);
}

// ---------------------------------------------------------------------------
// WMMA GEMM: D = act(A[MxK] * W[NxK]^T + bias)(+resid) -> f16 and/or f32.
// 128x128 block tile, BK=32, 8 waves x (2x4) 16x16 accumulators.
// Double-buffered LDS; tiles streamed via GLOBAL_LOAD_ASYNC_TO_LDS_B128.
// M,N multiples of 128; K multiple of 32 (true for all uses here).
// ---------------------------------------------------------------------------
__global__ __launch_bounds__(256)
void gemm_wmma(const h16* __restrict__ A, int lda,
               const h16* __restrict__ Wt,     // (N, K) row-major f16 weights
               const float* __restrict__ bias, // (N,)
               int M, int N, int K,
               h16* __restrict__ out16,
               float* __restrict__ out32,
               const float* __restrict__ residTok,   // (M,N) f32 token-major
               const float* __restrict__ residNCHW,  // NCHW f32 (h0 skip)
               int tokBase, int doGelu) {
  __shared__ __align__(16) h16 aS[2][128 * 32];
  __shared__ __align__(16) h16 wS[2][128 * 32];
  const int t = threadIdx.x, lane = t & 31, wave = t >> 5;
  const int l15 = lane & 15, hi = lane >> 4;
  const int wm = wave & 3, wn = wave >> 2;
  const int bm = blockIdx.x * 128, bn = blockIdx.y * 128;
  const int lr = t >> 1, lc = (t & 1) * 16;   // stage mapping: 2 threads/row

  const uint64_t aBase = (uint64_t)(uintptr_t)A;
  const uint64_t wBase = (uint64_t)(uintptr_t)Wt;
  const uint32_t ldsA = (uint32_t)(uintptr_t)(void*)&aS[0][0];
  const uint32_t ldsW = (uint32_t)(uintptr_t)(void*)&wS[0][0];

  auto stage = [&](int buf, int k0) {
    int ga = (int)(((size_t)(bm + lr) * lda + k0 + lc) * 2);
    uint32_t la = ldsA + (uint32_t)((buf * 4096 + lr * 32 + lc) * 2);
    async_b128(la,      aBase, ga);
    async_b128(la + 16, aBase, ga + 16);
    int gw = (int)(((size_t)(bn + lr) * K + k0 + lc) * 2);
    uint32_t lw = ldsW + (uint32_t)((buf * 4096 + lr * 32 + lc) * 2);
    async_b128(lw,      wBase, gw);
    async_b128(lw + 16, wBase, gw + 16);
  };

  v8f acc[2][4] = {};
  stage(0, 0);
  for (int k0 = 0; k0 < K; k0 += 32) {
    const int buf = (k0 >> 5) & 1;
    wait_async0();          // this wave's copies for `buf` are in LDS
    __syncthreads();        // everyone's copies landed; prev buffer fully read
    if (k0 + 32 < K) stage(buf ^ 1, k0 + 32);   // overlaps with compute below

    v16h af[2], bf[4];
    #pragma unroll
    for (int i = 0; i < 2; ++i)
      af[i] = frag_ld(&aS[buf][(wm * 32 + i * 16 + l15) * 32], hi);
    #pragma unroll
    for (int j = 0; j < 4; ++j)
      bf[j] = frag_ld(&wS[buf][(wn * 64 + j * 16 + l15) * 32], hi);
    #pragma unroll
    for (int i = 0; i < 2; ++i)
      #pragma unroll
      for (int j = 0; j < 4; ++j)
        acc[i][j] = wmma16(af[i], bf[j], acc[i][j]);
  }

  // Epilogue. C/D layout: elem r -> M = r + 8*hi ; N = lane&15.
  #pragma unroll
  for (int i = 0; i < 2; ++i)
    #pragma unroll
    for (int j = 0; j < 4; ++j)
      #pragma unroll
      for (int r = 0; r < 8; ++r) {
        int m = bm + wm * 32 + i * 16 + r + hi * 8;
        int n = bn + wn * 64 + j * 16 + l15;
        float v = acc[i][j][r] + bias[n];
        if (doGelu) v = 0.5f * v * (1.f + erff(v * 0.70710678f));
        if (residTok)  v += residTok[(size_t)m * N + n];
        if (residNCHW) {
          int gr = tokBase + m; int bb = gr >> 12, s = gr & (HW - 1);
          v += residNCHW[((size_t)bb * CC + n) * HW + s];
        }
        if (out32) out32[(size_t)m * N + n] = v;
        if (out16) out16[(size_t)m * N + n] = (h16)v;
      }
}

// ---------------------------------------------------------------------------
// Axial attention: one workgroup per (batch, head); dir 0 = rows(H), 1 = cols(W).
// seq len 64, head dim 1024 (16 channels x 64 inner positions).
// Head-dim traversal is reordered position-major (dd = p_local*16 + chl) so
// every (row, position) is a contiguous 16-half run -> async b128 staging.
// qkv: (tok,384) f16 (q|k|v); out: (tok,256) f16 at channel offset dir*128.
// ---------------------------------------------------------------------------
__global__ __launch_bounds__(256)
void attn_axial(const h16* __restrict__ qkv, h16* __restrict__ outA, int dir) {
  __shared__ float S[64 * 64];
  __shared__ __align__(16) h16 P[64 * 64];
  __shared__ __align__(16) h16 stage[4][64 * 64];   // 32KB ping-pong staging
  const int t = threadIdx.x, lane = t & 31, wave = t >> 5;
  const int l15 = lane & 15, hi = lane >> 4;
  const int bb = blockIdx.x >> 3, hd = blockIdx.x & 7;
  const int cbase = hd * 16;
  const uint64_t qbase = (uint64_t)(uintptr_t)qkv;
  const uint32_t lds0 = (uint32_t)(uintptr_t)(void*)&stage[0][0];

  // ---- Phase 1: S = Q K^T / 32 ----  (16 K-chunks of 64; Q/K double-buffered)
  auto stageQK = [&](int buf, int kc) {
    #pragma unroll
    for (int u = 0; u < 4; ++u) {
      int idx = u * 256 + t;          // 0..1023
      int mat = idx >> 9;             // 0 = Q, 1 = K
      int e = idx & 511;
      int r = e >> 3, pl = (e >> 1) & 3, hb = e & 1;
      int p = kc * 4 + pl;
      int tok = dir == 0 ? (bb * HW + r * 64 + p) : (bb * HW + p * 64 + r);
      int goff = (tok * 384 + cbase + mat * 128 + hb * 8) * 2;
      uint32_t loff = lds0 +
          (uint32_t)(((2 * buf + mat) * 4096 + r * 64 + pl * 16 + hb * 8) * 2);
      async_b128(loff, qbase, goff);
    }
  };
  const int ti0 = wave * 2, mi = ti0 >> 2, ni0 = ti0 & 3;
  v8f sacc[2] = {};
  stageQK(0, 0);
  for (int kc = 0; kc < 16; ++kc) {
    const int buf = kc & 1;
    wait_async0();
    __syncthreads();
    if (kc + 1 < 16) stageQK(buf ^ 1, kc + 1);
    const h16* qp = stage[2 * buf];
    const h16* kp = stage[2 * buf + 1];
    #pragma unroll
    for (int ks = 0; ks < 2; ++ks) {
      v16h aq = frag_ld(&qp[(mi * 16 + l15) * 64 + ks * 32], hi);
      #pragma unroll
      for (int i = 0; i < 2; ++i) {
        v16h bk = frag_ld(&kp[((ni0 + i) * 16 + l15) * 64 + ks * 32], hi);
        sacc[i] = wmma16(aq, bk, sacc[i]);
      }
    }
  }
  #pragma unroll
  for (int i = 0; i < 2; ++i)
    #pragma unroll
    for (int r = 0; r < 8; ++r) {
      int m = mi * 16 + r + hi * 8, n = (ni0 + i) * 16 + l15;
      S[m * 64 + n] = sacc[i][r] * 0.03125f;   // 1/sqrt(1024)
    }
  __syncthreads();

  if (t < 64) {       // softmax, one thread per row
    float mx = -1e30f;
    for (int j = 0; j < 64; ++j) mx = fmaxf(mx, S[t * 64 + j]);
    float sum = 0.f;
    for (int j = 0; j < 64; ++j) sum += expf(S[t * 64 + j] - mx);
    float inv = 1.f / sum;
    for (int j = 0; j < 64; ++j) P[t * 64 + j] = (h16)(expf(S[t * 64 + j] - mx) * inv);
  }
  __syncthreads();

  // ---- Phase 2: O = P V ----  (8 d-chunks of 128; V double-buffered)
  const int mi2 = wave >> 1, ns = (wave & 1) * 4;
  v16h ap[2];
  #pragma unroll
  for (int ks = 0; ks < 2; ++ks)
    ap[ks] = frag_ld(&P[(mi2 * 16 + l15) * 64 + ks * 32], hi);

  auto stageV = [&](int buf, int dc) {
    #pragma unroll
    for (int u = 0; u < 4; ++u) {
      int idx = u * 256 + t;          // 0..1023
      int r = idx >> 4, pl = (idx >> 1) & 7, hb = idx & 1;
      int p = dc * 8 + pl;
      int tok = dir == 0 ? (bb * HW + r * 64 + p) : (bb * HW + p * 64 + r);
      int goff = (tok * 384 + cbase + 256 + hb * 8) * 2;
      uint32_t loff = lds0 +
          (uint32_t)((buf * 8192 + r * 128 + pl * 16 + hb * 8) * 2);
      async_b128(loff, qbase, goff);
    }
  };
  stageV(0, 0);
  for (int dc = 0; dc < 8; ++dc) {
    const int buf = dc & 1;
    wait_async0();
    __syncthreads();
    if (dc + 1 < 8) stageV(buf ^ 1, dc + 1);
    const h16* Vr = &stage[0][0] + buf * 8192;   // row-major [j=64][dd=128]
    #pragma unroll
    for (int j = 0; j < 4; ++j) {
      const int dloc = (ns + j) * 16 + l15;
      v8f oa = {};
      #pragma unroll
      for (int ks = 0; ks < 2; ++ks) {
        v16h bv;
        #pragma unroll
        for (int e = 0; e < 16; ++e) {
          int kk = ks * 32 + e + (e & 8) + hi * 8;
          bv[e] = Vr[kk * 128 + dloc];
        }
        oa = wmma16(ap[ks], bv, oa);
      }
      #pragma unroll
      for (int r = 0; r < 8; ++r) {
        int m = mi2 * 16 + r + hi * 8;       // sequence position
        int chl = dloc & 15, p = dc * 8 + (dloc >> 4);
        int tok = dir == 0 ? (bb * HW + m * 64 + p) : (bb * HW + p * 64 + m);
        outA[(size_t)tok * CC + dir * 128 + cbase + chl] = (h16)oa[r];
      }
    }
  }
}

// ---------------------------------------------------------------------------
// Depthwise 3x3 PEG conv over token-major f32 (NHWC view)
// ---------------------------------------------------------------------------
__global__ __launch_bounds__(256)
void peg_kernel(const float* __restrict__ x, const float* __restrict__ w,
                const float* __restrict__ b, float* __restrict__ y) {
  size_t stride = (size_t)gridDim.x * blockDim.x;
  for (size_t idx = (size_t)blockIdx.x * blockDim.x + threadIdx.x;
       idx < (size_t)TOKS * CC; idx += stride) {
    int c = (int)(idx & 255);
    size_t tok = idx >> 8;
    int bb = (int)(tok >> 12), hh = (int)((tok >> 6) & 63), ww = (int)(tok & 63);
    float acc = b[c];
    #pragma unroll
    for (int dy = -1; dy <= 1; ++dy)
      #pragma unroll
      for (int dx = -1; dx <= 1; ++dx) {
        int h2 = hh + dy, w2 = ww + dx;
        if ((unsigned)h2 < 64u && (unsigned)w2 < 64u)
          acc += w[c * 9 + (dy + 1) * 3 + (dx + 1)] *
                 x[((size_t)bb * HW + h2 * 64 + w2) * CC + c];
      }
    y[idx] = acc;
  }
}

__global__ __launch_bounds__(256)
void pool_kernel(const float* __restrict__ y, float* __restrict__ avg,
                 float* __restrict__ mx) {
  int bc = blockIdx.x, bb = bc >> 8, c = bc & 255;
  const float* p = y + (size_t)bb * HW * CC + c;
  float s = 0.f, m = -1e30f;
  for (int i = threadIdx.x; i < HW; i += 256) {
    float v = p[(size_t)i * CC]; s += v; m = fmaxf(m, v);
  }
  __shared__ float ss[256], sm[256];
  ss[threadIdx.x] = s; sm[threadIdx.x] = m;
  __syncthreads();
  for (int o = 128; o > 0; o >>= 1) {
    if (threadIdx.x < o) {
      ss[threadIdx.x] += ss[threadIdx.x + o];
      sm[threadIdx.x] = fmaxf(sm[threadIdx.x], sm[threadIdx.x + o]);
    }
    __syncthreads();
  }
  if (threadIdx.x == 0) { avg[bc] = ss[0] * (1.f / HW); mx[bc] = sm[0]; }
}

// Squeeze-excite gate: only the 3x3 center tap contributes (1x1 spatial, pad 1).
__global__ __launch_bounds__(256)
void ca_gate_kernel(const float* __restrict__ avg, const float* __restrict__ mx,
                    const float* __restrict__ w1, const float* __restrict__ w2,
                    float* __restrict__ gate) {
  int bb = blockIdx.x, t = threadIdx.x;
  __shared__ float ta[32], tm[32];
  if (t < 32) {
    float sa = 0.f, sm = 0.f;
    for (int ci = 0; ci < 256; ++ci) {
      float w = w1[(t * 256 + ci) * 9 + 4];
      sa += w * avg[bb * 256 + ci];
      sm += w * mx[bb * 256 + ci];
    }
    ta[t] = fmaxf(sa, 0.f); tm[t] = fmaxf(sm, 0.f);
  }
  __syncthreads();
  float ga = 0.f, gm = 0.f;
  for (int j = 0; j < 32; ++j) {
    float w = w2[(t * 32 + j) * 9 + 4];
    ga += w * ta[j]; gm += w * tm[j];
  }
  gate[bb * 256 + t] = 1.f / (1.f + expf(-(ga + gm)));
}

__global__ __launch_bounds__(256)
void final_kernel(const float* __restrict__ y, const float* __restrict__ x,
                  const float* __restrict__ gate, float* __restrict__ out) {
  size_t stride = (size_t)gridDim.x * blockDim.x;
  for (size_t idx = (size_t)blockIdx.x * blockDim.x + threadIdx.x;
       idx < (size_t)16 * CC * HW; idx += stride) {
    int s = (int)(idx & (HW - 1));
    size_t r = idx >> 12;
    int c = (int)(r & 255), bb = (int)(r >> 8);
    size_t tok = (size_t)bb * HW + s;
    out[idx] = y[tok * CC + c] * gate[bb * 256 + c] + x[tok * CC + c];
  }
}

// ---------------------------------------------------------------------------
extern "C" void kernel_launch(void* const* d_in, const int* in_sizes, int n_in,
                              void* d_out, int out_size, void* d_ws, size_t ws_size,
                              hipStream_t stream) {
  (void)in_sizes; (void)n_in; (void)out_size; (void)ws_size;
  const float* x      = (const float*)d_in[0];
  const float* ln1_g  = (const float*)d_in[1];
  const float* ln1_b  = (const float*)d_in[2];
  const float* cin_w  = (const float*)d_in[3];
  const float* cin_b  = (const float*)d_in[4];
  const float* ch_w   = (const float*)d_in[5];
  const float* ch_b   = (const float*)d_in[6];
  const float* cv_w   = (const float*)d_in[7];
  const float* cv_b   = (const float*)d_in[8];
  const float* fuse_w = (const float*)d_in[9];
  const float* fuse_b = (const float*)d_in[10];
  const float* ln2_g  = (const float*)d_in[11];
  const float* ln2_b  = (const float*)d_in[12];
  const float* fc1_w  = (const float*)d_in[13];
  const float* fc1_b  = (const float*)d_in[14];
  const float* fc2_w  = (const float*)d_in[15];
  const float* fc2_b  = (const float*)d_in[16];
  const float* peg_w  = (const float*)d_in[17];
  const float* peg_b  = (const float*)d_in[18];
  const float* ca1_w  = (const float*)d_in[19];
  const float* ca2_w  = (const float*)d_in[20];

  char* ws = (char*)d_ws;
  const size_t MB = 1024 * 1024;
  h16*   xn16   = (h16*)(ws + 0);          // 32MB; reused for LN2 output
  h16*   qkvh   = (h16*)(ws + 32 * MB);    // 48MB
  h16*   qkvv   = (h16*)(ws + 80 * MB);    // 48MB
  h16*   xi16   = (h16*)(ws + 128 * MB);   // 32MB
  h16*   attn16 = (h16*)(ws + 160 * MB);   // 32MB; reused as fc1 chunk buffer
  float* x2     = (float*)(ws + 192 * MB); // 64MB; reused as PEG output y
  float* x3     = (float*)(ws + 256 * MB); // 64MB
  float* avg    = (float*)(ws + 320 * MB);
  float* mxp    = (float*)(ws + 320 * MB + 64 * 1024);
  float* gate   = (float*)(ws + 320 * MB + 128 * 1024);
  h16*   wpool  = (h16*)(ws + 321 * MB);   // f16 weight pool (~1.5MB)
  float* out    = (float*)d_out;

  // f16 weight pool offsets (halves)
  h16* wCIN = wpool + 0;        // 256*256
  h16* wCH  = wpool + 65536;    // 384*128
  h16* wCV  = wpool + 114688;   // 384*128
  h16* wFU  = wpool + 163840;   // 256*256
  h16* wF1  = wpool + 229376;   // 1024*256
  h16* wF2  = wpool + 491520;   // 256*1024

  cvt_f32_to_f16<<<65536 / 256, 256, 0, stream>>>(cin_w,  wCIN, 65536);
  cvt_f32_to_f16<<<49152 / 256, 256, 0, stream>>>(ch_w,   wCH,  49152);
  cvt_f32_to_f16<<<49152 / 256, 256, 0, stream>>>(cv_w,   wCV,  49152);
  cvt_f32_to_f16<<<65536 / 256, 256, 0, stream>>>(fuse_w, wFU,  65536);
  cvt_f32_to_f16<<<262144 / 256, 256, 0, stream>>>(fc1_w, wF1,  262144);
  cvt_f32_to_f16<<<262144 / 256, 256, 0, stream>>>(fc2_w, wF2,  262144);

  // 1) LN1 (NCHW f32 -> token-major f16)
  ln_nchw_to_tok16<<<TOKS / 256, 256, 0, stream>>>(x, ln1_g, ln1_b, xn16);
  // 2) conv_in: 65536x256 @ 256x256
  gemm_wmma<<<dim3(512, 2), 256, 0, stream>>>(xn16, 256, wCIN, cin_b,
      TOKS, 256, 256, xi16, nullptr, nullptr, nullptr, 0, 0);
  // 3) QKV 1x1 convs on each 128-ch half: 65536x128 @ 384x128
  gemm_wmma<<<dim3(512, 3), 256, 0, stream>>>(xi16, 256, wCH, ch_b,
      TOKS, 384, 128, qkvh, nullptr, nullptr, nullptr, 0, 0);
  gemm_wmma<<<dim3(512, 3), 256, 0, stream>>>(xi16 + 128, 256, wCV, cv_b,
      TOKS, 384, 128, qkvv, nullptr, nullptr, nullptr, 0, 0);
  // 4) Axial attention: 128 (b,head) problems per direction
  attn_axial<<<128, 256, 0, stream>>>(qkvh, attn16, 0);
  attn_axial<<<128, 256, 0, stream>>>(qkvv, attn16, 1);
  // 5) fuse conv + h0 residual (NCHW read) -> x2 f32 token-major
  gemm_wmma<<<dim3(512, 2), 256, 0, stream>>>(attn16, 256, wFU, fuse_b,
      TOKS, 256, 256, nullptr, x2, nullptr, x, 0, 0);
  // 6) LN2 -> f16 (reuse xn16 slot)
  ln_tok16<<<TOKS / 256, 256, 0, stream>>>(x2, ln2_g, ln2_b, xn16);
  // 7) MLP, chunked over 4 token blocks (fc1 buffer reuses attn16 slot)
  h16* fc1buf = attn16;
  for (int mc = 0; mc < 4; ++mc) {
    size_t t0 = (size_t)mc * 16384;
    gemm_wmma<<<dim3(128, 8), 256, 0, stream>>>(xn16 + t0 * 256, 256,
        wF1, fc1_b, 16384, 1024, 256, fc1buf, nullptr, nullptr, nullptr, 0, 1);
    gemm_wmma<<<dim3(128, 2), 256, 0, stream>>>(fc1buf, 1024,
        wF2, fc2_b, 16384, 256, 1024, nullptr, x3 + t0 * 256,
        x2 + t0 * 256, nullptr, 0, 0);
  }
  // 8) PEG depthwise 3x3 (y reuses x2 slot), pooling, gate, final blend
  float* y32 = x2;
  peg_kernel<<<8192, 256, 0, stream>>>(x3, peg_w, peg_b, y32);
  pool_kernel<<<4096, 256, 0, stream>>>(y32, avg, mxp);
  ca_gate_kernel<<<16, 256, 0, stream>>>(avg, mxp, ca1_w, ca2_w, gate);
  final_kernel<<<8192, 256, 0, stream>>>(y32, x3, gate, out);
}
